// SALAD_16037407883769
// MI455X (gfx1250) — matrix operand order repaired and checked by reading, not compiled, not run on hardware
//
#include <hip/hip_runtime.h>

// ---------------------------------------------------------------------------
// Types for CDNA5 WMMA
// ---------------------------------------------------------------------------
typedef __attribute__((ext_vector_type(16))) __bf16 v16bf;
typedef __attribute__((ext_vector_type(8)))  float  v8f;

union ABFrag { unsigned u[8]; v16bf v; };

static __device__ __forceinline__ unsigned short f32_to_bf16(float f) {
    unsigned u = __builtin_bit_cast(unsigned, f);
    unsigned rounding = 0x7FFFu + ((u >> 16) & 1u);   // round-to-nearest-even
    u += rounding;
    return (unsigned short)(u >> 16);
}
static __device__ __forceinline__ unsigned pack_bf16(float lo, float hi) {
    return (unsigned)f32_to_bf16(lo) | ((unsigned)f32_to_bf16(hi) << 16);
}
// Low 32 bits of a generic pointer to __shared__ = LDS byte offset on amdgcn.
static __device__ __forceinline__ unsigned lds_offset(const void* p) {
    return (unsigned)(unsigned long long)p;
}

// ---------------------------------------------------------------------------
// 1) Weight prep: fp32 -> bf16; stack cf_w1/sc_w1 into one [1024][1536]
// ---------------------------------------------------------------------------
__global__ void prep_weights(const float* __restrict__ cf_w1, const float* __restrict__ cf_b1,
                             const float* __restrict__ sc_w1, const float* __restrict__ sc_b1,
                             const float* __restrict__ cf_w2, const float* __restrict__ sc_w2,
                             unsigned short* __restrict__ Wc1, float* __restrict__ b1c,
                             unsigned short* __restrict__ W2cf, unsigned short* __restrict__ W2sc)
{
    const long n1 = 1024L * 1536, n2 = 128L * 512, n3 = 64L * 512;
    long i = (long)blockIdx.x * blockDim.x + threadIdx.x;
    if (i < n1) {
        long o = i / 1536, k = i % 1536;
        float val = (o < 512) ? cf_w1[o * 1536 + k] : sc_w1[(o - 512) * 1536 + k];
        Wc1[i] = f32_to_bf16(val);
    } else if (i < n1 + n2) {
        long j = i - n1;
        W2cf[j] = f32_to_bf16(cf_w2[j]);
    } else if (i < n1 + n2 + n3) {
        long j = i - n1 - n2;
        W2sc[j] = f32_to_bf16(sc_w2[j]);
    }
    if (i < 1024) b1c[i] = (i < 512) ? cf_b1[i] : sc_b1[i - 512];
}

// ---------------------------------------------------------------------------
// 2) Tiled bf16 WMMA GEMM:  Out[b][o][n] = act( sum_k W[o][k]*In[b][k][n] + bias[o] )
//    N fixed at 1024. Block tile 64(O) x 128(N), K-step 32, 8 waves (4x2).
//    A tile (bf16 weights) staged with GLOBAL_LOAD_ASYNC_TO_LDS_B128 (ASYNCcnt);
//    B tile staged via VALU (needs f32->bf16 cvt + pair-transpose).
// ---------------------------------------------------------------------------
template<bool IN_F32, bool RELU, bool OUT_BF16>
__global__ __launch_bounds__(256)
void wmma_gemm(const unsigned short* __restrict__ Wb,   // [O][K] bf16
               const float* __restrict__ bias,          // [O]
               const void* __restrict__ In,             // [B][K][1024] f32 or bf16
               long inBatchStride, long inRowOffset,    // elements
               void* __restrict__ Out,                  // [B][O][1024]
               int O, int K)
{
    const int N = 1024;
    __shared__ unsigned aT[64 * 20];    // [row][kpair], stride 20 => 16B-aligned 4-uint chunks
    __shared__ unsigned bT[128 * 17];   // [col][kpair]

    const int tid  = threadIdx.x;
    const int lane = tid & 31;
    const int wave = tid >> 5;
    const int wm = wave & 3;            // 4 waves along O
    const int wn = wave >> 2;           // 2 waves along N
    const int b     = blockIdx.z;
    const int oBase = blockIdx.y * 64;
    const int nBase = blockIdx.x * 128;

    const float*          inF = IN_F32 ? ((const float*)In          + (long)b * inBatchStride + inRowOffset) : nullptr;
    const unsigned short* inH = IN_F32 ? nullptr : ((const unsigned short*)In + (long)b * inBatchStride + inRowOffset);

    // Per-thread fixed staging coordinates
    const int ar  = tid >> 2;             // A row 0..63
    const int adp = (tid & 3) * 4;        // A pair index 0,4,8,12 (16B chunk)
    const unsigned aLds = lds_offset(&aT[ar * 20 + adp]);
    const unsigned short* aGlob = Wb + (long)(oBase + ar) * K + adp * 2;

    const v8f zf = {0.f,0.f,0.f,0.f,0.f,0.f,0.f,0.f};
    v8f acc[4] = {zf, zf, zf, zf};

    for (int k0 = 0; k0 < K; k0 += 32) {
        // ---- stage A: async DMA global(bf16) -> LDS, 16B per lane ----
        {
            const unsigned short* gp = aGlob + k0;   // 16B-aligned
            asm volatile("global_load_async_to_lds_b128 %0, %1, off"
                         :: "v"(aLds), "v"(gp) : "memory");
        }
        // ---- stage B: 128 cols x 16 pairs, pair-packed per column (VALU cvt) ----
        {
            int n  = tid & 127;
            int p0 = tid >> 7;                 // 0 or 1
            for (int p = p0; p < 16; p += 2) {
                int k = k0 + 2 * p;
                unsigned pr;
                if (IN_F32) {
                    float x0 = inF[(long)k * N + nBase + n];
                    float x1 = inF[(long)(k + 1) * N + nBase + n];
                    pr = pack_bf16(x0, x1);
                } else {
                    unsigned short h0 = inH[(long)k * N + nBase + n];
                    unsigned short h1 = inH[(long)(k + 1) * N + nBase + n];
                    pr = (unsigned)h0 | ((unsigned)h1 << 16);
                }
                bT[n * 17 + p] = pr;
            }
        }
        asm volatile("s_wait_asynccnt 0x0" ::: "memory");
        __syncthreads();

        // ---- A fragment (16-bit A layout: lanes 0-15 K=0..7,16..23; 16-31 +8) ----
        ABFrag af;
        {
            int m  = wm * 16 + (lane & 15);
            int kh = (lane < 16) ? 0 : 4;      // pair offset
            #pragma unroll
            for (int v = 0; v < 8; ++v) {
                int kp = ((v < 4) ? v : 8 + (v - 4)) + kh;
                af.u[v] = aT[m * 20 + kp];
            }
        }
        // ---- 4 WMMAs per wave over its 64 columns ----
        #pragma unroll
        for (int nt = 0; nt < 4; ++nt) {
            ABFrag bf;
            int n  = wn * 64 + nt * 16 + (lane & 15);
            int kg = (lane < 16) ? 0 : 4;      // pair offset (K halves across lane groups)
            #pragma unroll
            for (int v = 0; v < 8; ++v)
                bf.u[v] = bT[n * 17 + kg + v + (v & 4)];
            acc[nt] = __builtin_amdgcn_wmma_f32_16x16x32_bf16(
                false, af.v, false, bf.v, (short)0, acc[nt], false, false);
        }
        __syncthreads();
    }

    // ---- epilogue: bias (+relu), store ----
    #pragma unroll
    for (int nt = 0; nt < 4; ++nt) {
        int n = nBase + wn * 64 + nt * 16 + (lane & 15);
        #pragma unroll
        for (int r = 0; r < 8; ++r) {
            int m = oBase + wm * 16 + r + ((lane < 16) ? 0 : 8);
            float val = acc[nt][r] + bias[m];
            if (RELU) val = fmaxf(val, 0.f);
            long off = (long)b * O * N + (long)m * N + n;
            if (OUT_BF16) ((unsigned short*)Out)[off] = f32_to_bf16(val);
            else          ((float*)Out)[off] = val;
        }
    }
}

// ---------------------------------------------------------------------------
// 3) Token MLP: tk = relu(t @ w1^T + b1) @ w2^T + b2   (tiny, fp32 VALU)
// ---------------------------------------------------------------------------
__global__ __launch_bounds__(256)
void tk_mlp(const float* __restrict__ t,
            const float* __restrict__ w1, const float* __restrict__ b1,
            const float* __restrict__ w2, const float* __restrict__ b2,
            float* __restrict__ Tk)
{
    __shared__ float tr[1536];
    __shared__ float hid[512];
    int b = blockIdx.x, tid = threadIdx.x;
    for (int i = tid; i < 1536; i += 256) tr[i] = t[b * 1536 + i];
    __syncthreads();
    for (int j = tid; j < 512; j += 256) {
        float s = b1[j];
        const float* wr = w1 + (long)j * 1536;
        for (int k = 0; k < 1536; ++k) s += wr[k] * tr[k];
        hid[j] = fmaxf(s, 0.f);
    }
    __syncthreads();
    {
        int g = tid;   // 256 outputs, 256 threads
        float s = b2[g];
        const float* wr = w2 + (long)g * 512;
        for (int h = 0; h < 512; ++h) s += wr[h] * hid[h];
        Tk[b * 256 + g] = s;
    }
}

// ---------------------------------------------------------------------------
// 4) Log-space Sinkhorn (3 iters) + exp -> P bf16 [B][64][1024]
//    One block per batch, 1024 threads (32 waves, wave32 shuffles).
// ---------------------------------------------------------------------------
__global__ __launch_bounds__(1024)
void sinkhorn(const float* __restrict__ S,        // [B][64][1024]
              const float* __restrict__ alphaPtr,
              unsigned short* __restrict__ P)     // [B][64][1024] bf16
{
    __shared__ float u[65];
    __shared__ float v[1024];
    const int b = blockIdx.x, tid = threadIdx.x;
    const int lane = tid & 31, wave = tid >> 5;
    const float alpha  = alphaPtr[0];
    const float norm   = -logf(1088.0f);           // -log(m+n), m=64, n=1024
    const float lm_bin = logf(960.0f) + norm;      // log(n-m)+norm
    const float* Sb = S + (long)b * 64 * 1024;

    v[tid] = 0.f;
    __syncthreads();

    for (int it = 0; it < 3; ++it) {
        // u[m] = log_mu[m] - LSE_n(Z[m][n] + v[n]); one wave per row
        for (int m = wave; m < 65; m += 32) {
            float mx = -INFINITY, sm = 0.f;
            for (int n = lane; n < 1024; n += 32) {
                float z = ((m < 64) ? Sb[m * 1024 + n] : alpha) + v[n];
                if (z > mx) { sm = sm * expf(mx - z) + 1.f; mx = z; }
                else        { sm += expf(z - mx); }
            }
            #pragma unroll
            for (int off = 16; off > 0; off >>= 1) {
                float omx = __shfl_xor(mx, off, 32);
                float osm = __shfl_xor(sm, off, 32);
                float M = fmaxf(mx, omx);
                sm = sm * expf(mx - M) + osm * expf(omx - M);
                mx = M;
            }
            if (lane == 0) u[m] = ((m < 64) ? norm : lm_bin) - (mx + logf(sm));
        }
        __syncthreads();
        // v[n] = log_nu - LSE_m(Z[m][n] + u[m]); one thread per column
        {
            int n = tid;
            float mx = -INFINITY, sm = 0.f;
            for (int m = 0; m < 65; ++m) {
                float z = ((m < 64) ? Sb[m * 1024 + n] : alpha) + u[m];
                if (z > mx) { sm = sm * expf(mx - z) + 1.f; mx = z; }
                else        { sm += expf(z - mx); }
            }
            v[n] = norm - (mx + logf(sm));
        }
        __syncthreads();
    }

    unsigned short* Pb = P + (long)b * 64 * 1024;
    for (int m = 0; m < 64; ++m) {
        float z = Sb[m * 1024 + tid] + u[m] + v[tid] - norm;
        Pb[m * 1024 + tid] = f32_to_bf16(expf(z));
    }
}

// ---------------------------------------------------------------------------
// 5) VLAD aggregation: agg[b][l][m] = sum_n f[b][l][n] * P[b][m][n]
//    NT layout -> both fragments load directly from global (contiguous pairs).
//    One wave computes a 16x16 tile; 1024 tiles total; 128 blocks x 8 waves.
// ---------------------------------------------------------------------------
__global__ __launch_bounds__(256)
void agg_wmma(const unsigned short* __restrict__ F,   // [B][128][1024] bf16
              const unsigned short* __restrict__ P,   // [B][64][1024] bf16
              float* __restrict__ Agg)                // [B][128][64]
{
    const int lane = threadIdx.x & 31;
    const int wave = threadIdx.x >> 5;
    const int tile = blockIdx.x * 8 + wave;   // 0..1023 = B(32) * lt(8) * mt(4)
    const int b  = tile >> 5;
    const int lt = (tile >> 2) & 7;
    const int mt = tile & 3;

    const unsigned* Fb = (const unsigned*)(F + (long)b * 128 * 1024);  // pair view
    const unsigned* Pb = (const unsigned*)(P + (long)b * 64 * 1024);

    const v8f zf = {0.f,0.f,0.f,0.f,0.f,0.f,0.f,0.f};
    v8f acc = zf;

    const int lrow = lt * 16 + (lane & 15);
    const int mrow = mt * 16 + (lane & 15);
    const int kh_a = (lane < 16) ? 0 : 4;     // A pair offset
    const int kg_b = (lane < 16) ? 0 : 4;     // B pair offset

    for (int k0 = 0; k0 < 1024; k0 += 32) {
        ABFrag af, bf;
        int kp0 = k0 >> 1;
        #pragma unroll
        for (int v = 0; v < 8; ++v) {
            int kpa = kp0 + ((v < 4) ? v : 8 + (v - 4)) + kh_a;
            af.u[v] = Fb[lrow * 512 + kpa];
            bf.u[v] = Pb[mrow * 512 + kp0 + kg_b + v + (v & 4)];
        }
        acc = __builtin_amdgcn_wmma_f32_16x16x32_bf16(
            false, af.v, false, bf.v, (short)0, acc, false, false);
    }

    const int mcol = mt * 16 + (lane & 15);
    #pragma unroll
    for (int r = 0; r < 8; ++r) {
        int l = lt * 16 + r + ((lane < 16) ? 0 : 8);
        Agg[(long)b * 8192 + l * 64 + mcol] = acc[r];
    }
}

// ---------------------------------------------------------------------------
// 6) Finalize: per-column l2norm of agg, l2norm of tk, concat, global l2norm
// ---------------------------------------------------------------------------
__global__ __launch_bounds__(256)
void finalize(const float* __restrict__ Tk, const float* __restrict__ Agg,
              float* __restrict__ out)
{
    __shared__ float vtk[256];
    __shared__ float coln[64];
    __shared__ float red[256];
    const int b = blockIdx.x, tid = threadIdx.x;
    const int lane = tid & 31, wave = tid >> 5;
    const float* Ab = Agg + (long)b * 8192;
    const float EPS = 1e-12f;

    // tk l2norm
    float tv = Tk[b * 256 + tid];
    float ss = tv * tv;
    #pragma unroll
    for (int off = 16; off > 0; off >>= 1) ss += __shfl_xor(ss, off, 32);
    if (lane == 0) red[wave] = ss;
    __syncthreads();
    if (tid == 0) { float s = 0.f; for (int w = 0; w < 8; ++w) s += red[w]; red[0] = s; }
    __syncthreads();
    float tknorm = fmaxf(sqrtf(red[0]), EPS);
    vtk[tid] = tv / tknorm;
    __syncthreads();

    // per-cluster-column sumsq over l (axis of length 128)
    {
        int m = tid & 63, seg = tid >> 6;
        float p = 0.f;
        for (int l = seg * 32; l < seg * 32 + 32; ++l) {
            float a = Ab[l * 64 + m];
            p += a * a;
        }
        red[tid] = p;
    }
    __syncthreads();
    if (tid < 64) {
        float css = red[tid] + red[64 + tid] + red[128 + tid] + red[192 + tid];
        float cn = fmaxf(sqrtf(css), EPS);
        coln[tid] = cn;
        red[tid] = css / (cn * cn);          // normalized column's sumsq contribution
    }
    __syncthreads();

    // total sumsq of concatenated normalized vector
    float part = vtk[tid] * vtk[tid] + ((tid < 64) ? red[tid] : 0.f);
    #pragma unroll
    for (int off = 16; off > 0; off >>= 1) part += __shfl_xor(part, off, 32);
    __syncthreads();
    if (lane == 0) red[wave] = part;
    __syncthreads();
    if (tid == 0) { float s = 0.f; for (int w = 0; w < 8; ++w) s += red[w]; red[0] = s; }
    __syncthreads();
    float tot = fmaxf(sqrtf(red[0]), EPS);

    float* ob = out + (long)b * 8448;
    ob[tid] = vtk[tid] / tot;
    for (int idx = tid; idx < 8192; idx += 256) {
        int m = idx & 63;
        ob[256 + idx] = (Ab[idx] / coln[m]) / tot;
    }
}

// ---------------------------------------------------------------------------
// Launch
// ---------------------------------------------------------------------------
extern "C" void kernel_launch(void* const* d_in, const int* in_sizes, int n_in,
                              void* d_out, int out_size, void* d_ws, size_t ws_size,
                              hipStream_t stream) {
    const float* x      = (const float*)d_in[0];
    const float* t      = (const float*)d_in[1];
    const float* cf_w1  = (const float*)d_in[2];
    const float* cf_b1  = (const float*)d_in[3];
    const float* cf_w2  = (const float*)d_in[4];
    const float* cf_b2  = (const float*)d_in[5];
    const float* sc_w1  = (const float*)d_in[6];
    const float* sc_b1  = (const float*)d_in[7];
    const float* sc_w2  = (const float*)d_in[8];
    const float* sc_b2  = (const float*)d_in[9];
    const float* tk_w1  = (const float*)d_in[10];
    const float* tk_b1  = (const float*)d_in[11];
    const float* tk_w2  = (const float*)d_in[12];
    const float* tk_b2  = (const float*)d_in[13];
    const float* alpha  = (const float*)d_in[14];

    char* ws = (char*)d_ws;
    // byte offsets (256-aligned)
    const size_t OFF_WC1  = 0;                         // 1024*1536*2 = 3,145,728
    const size_t OFF_B1C  = OFF_WC1  + 3145728;        // 4096
    const size_t OFF_W2CF = OFF_B1C  + 4096;           // 128*512*2 = 131072
    const size_t OFF_W2SC = OFF_W2CF + 131072;         // 64*512*2  = 65536
    const size_t OFF_H    = OFF_W2SC + 65536;          // 32*1024*1024*2 = 64 MB
    const size_t OFF_F    = OFF_H    + 67108864;       // 32*128*1024*2 = 8 MB
    const size_t OFF_S    = OFF_F    + 8388608;        // 32*64*1024*4  = 8 MB
    const size_t OFF_P    = OFF_S    + 8388608;        // 32*64*1024*2  = 4 MB
    const size_t OFF_TK   = OFF_P    + 4194304;        // 32*256*4
    const size_t OFF_AGG  = OFF_TK   + 32768;          // 32*128*64*4 = 1 MB

    unsigned short* Wc1  = (unsigned short*)(ws + OFF_WC1);
    float*          b1c  = (float*)(ws + OFF_B1C);
    unsigned short* W2cf = (unsigned short*)(ws + OFF_W2CF);
    unsigned short* W2sc = (unsigned short*)(ws + OFF_W2SC);
    unsigned short* Hbuf = (unsigned short*)(ws + OFF_H);
    unsigned short* Fbuf = (unsigned short*)(ws + OFF_F);
    float*          Sbuf = (float*)(ws + OFF_S);
    unsigned short* Pbuf = (unsigned short*)(ws + OFF_P);
    float*          Tkb  = (float*)(ws + OFF_TK);
    float*          Aggb = (float*)(ws + OFF_AGG);

    // 1) weight conversion (1024*1536 + 128*512 + 64*512 elements)
    prep_weights<<<6528, 256, 0, stream>>>(cf_w1, cf_b1, sc_w1, sc_b1, cf_w2, sc_w2,
                                           Wc1, b1c, W2cf, W2sc);

    // 2) layer-1 GEMM (stacked cf+sc): H = relu(Wc1 @ x + b1c), bf16 out
    dim3 g1(1024 / 128, 1024 / 64, 32);
    wmma_gemm<true, true, true><<<g1, 256, 0, stream>>>(
        Wc1, b1c, x, 1536L * 1024, 0, Hbuf, 1024, 1536);

    // 3) layer-2 cf: f = cf_w2 @ H[0:512] + cf_b2, bf16 out
    dim3 g2(1024 / 128, 128 / 64, 32);
    wmma_gemm<false, false, true><<<g2, 256, 0, stream>>>(
        W2cf, cf_b2, Hbuf, 1024L * 1024, 0, Fbuf, 128, 512);

    // 4) layer-2 sc: scores = sc_w2 @ H[512:1024] + sc_b2, fp32 out
    dim3 g3(1024 / 128, 64 / 64, 32);
    wmma_gemm<false, false, false><<<g3, 256, 0, stream>>>(
        W2sc, sc_b2, Hbuf, 1024L * 1024, 512L * 1024, Sbuf, 64, 512);

    // 5) token MLP
    tk_mlp<<<32, 256, 0, stream>>>(t, tk_w1, tk_b1, tk_w2, tk_b2, Tkb);

    // 6) Sinkhorn OT -> P (bf16)
    sinkhorn<<<32, 1024, 0, stream>>>(Sbuf, alpha, Pbuf);

    // 7) VLAD aggregation: agg = f @ P^T (WMMA, direct global fragments)
    agg_wmma<<<128, 256, 0, stream>>>(Fbuf, Pbuf, Aggb);

    // 8) normalize + concat + normalize
    finalize<<<32, 256, 0, stream>>>(Tkb, Aggb, (float*)d_out);
}
